// MinimalLRU_8332236554552
// MI455X (gfx1250) — compile-verified, matrix operand order checked by reading, not run on hardware
//
#include <hip/hip_runtime.h>
#include <hip/hip_bf16.h>
#include <math.h>
#include <stdint.h>

typedef __attribute__((ext_vector_type(16))) _Float16 v16h;
typedef __attribute__((ext_vector_type(8)))  _Float16 v8h;
typedef __attribute__((ext_vector_type(8)))  float    v8f;

#define TOKEN_DIM 512
#define TWO_H     256
#define HDIM      128
#define STATE_DIM 256
#define BATCH     8
#define TSEQ      8192
#define NTOK      (BATCH * TSEQ)          // 65536
#define CHUNK     64
#define NCHUNK    (TSEQ / CHUNK)          // 128
#define NSUM      (BATCH * NCHUNK * HDIM) // 131072

#define USE_ASYNC_STAGE 1

// ---------- CDNA5 async global->LDS staging (ASYNCcnt path) ----------
// global_load_async_to_lds_b128: VDST = per-lane LDS byte address,
// VADDR = 64-bit global address (GV mode).  VDST is an offset from the LDS base.
__device__ __forceinline__ void async_copy16(uint32_t lds_off, uint64_t gaddr) {
  asm volatile("global_load_async_to_lds_b128 %0, %1, off"
               :: "v"(lds_off), "v"(gaddr) : "memory");
}
__device__ __forceinline__ void wait_async0() {
  asm volatile("s_wait_asynccnt 0x0" ::: "memory");
}
// Generic shared pointer low 32 bits == LDS byte offset (ISA 10.2 aperture map)
__device__ __forceinline__ uint32_t lds_off_of(const void* p) {
  return (uint32_t)(uintptr_t)p;
}

// ---------- fragment loaders (CDNA5 WMMA 16x16x32 f16 layouts) ----------
// A (16xK=32, row-major mem, K contiguous): lane L (x=L&15, hi=L>>4) holds
// elements 0..7  = K (hi?8:0)..+7   and elements 8..15 = K 16+(hi?8:0)..+7.
// B (32x16) uses the same per-lane K-blocking with lane&15 = N column.

__device__ __forceinline__ v16h frag_from_f32(const float* __restrict__ p) {
  const float4 f0 = ((const float4*)p)[0];
  const float4 f1 = ((const float4*)p)[1];
  const float4 f2 = ((const float4*)(p + 16))[0];
  const float4 f3 = ((const float4*)(p + 16))[1];
  v16h r;
  r[0]=(_Float16)f0.x;  r[1]=(_Float16)f0.y;  r[2]=(_Float16)f0.z;  r[3]=(_Float16)f0.w;
  r[4]=(_Float16)f1.x;  r[5]=(_Float16)f1.y;  r[6]=(_Float16)f1.z;  r[7]=(_Float16)f1.w;
  r[8]=(_Float16)f2.x;  r[9]=(_Float16)f2.y;  r[10]=(_Float16)f2.z; r[11]=(_Float16)f2.w;
  r[12]=(_Float16)f3.x; r[13]=(_Float16)f3.y; r[14]=(_Float16)f3.z; r[15]=(_Float16)f3.w;
  return r;
}

__device__ __forceinline__ v16h frag_from_h16(const _Float16* __restrict__ p) {
  const v8h lo = *(const v8h*)p;
  const v8h hp = *(const v8h*)(p + 16);
  v16h r;
#pragma unroll
  for (int i = 0; i < 8; ++i) { r[i] = lo[i]; r[8 + i] = hp[i]; }
  return r;
}

__device__ __forceinline__ void lam_of(int h, const float* __restrict__ log_radius,
                                       const float* __restrict__ phase,
                                       float& lre, float& lim) {
  const float rad = 1.0f / (1.0f + __expf(-log_radius[h]));
  lre = rad * __cosf(phase[h]);
  lim = rad * __sinf(phase[h]);
}

// =======================================================================
// K0: one-time f32 -> f16 conversion of W_in (256x512) and W_out (256x256)
// =======================================================================
__global__ __launch_bounds__(256) void k0_cvt_weights(const float* __restrict__ W_in,
                                                      const float* __restrict__ W_out,
                                                      _Float16* __restrict__ Wh_in,
                                                      _Float16* __restrict__ Wh_out) {
  const int i = blockIdx.x * 256 + threadIdx.x;          // 0 .. 196607
  if (i < TWO_H * TOKEN_DIM) {
    Wh_in[i] = (_Float16)W_in[i];
  } else {
    const int j = i - TWO_H * TOKEN_DIM;                 // 0 .. 65535
    Wh_out[j] = (_Float16)W_out[j];
  }
}

// =======================================================================
// K1: inp[bt][n] = sum_k x[bt][k] * W_in[n][k] + b_in[n]    (f16 WMMA)
// block = 256 thr (8 waves), tile M=128 (16 rows/wave), N=256, K streamed 32
// =======================================================================
__global__ __launch_bounds__(256) void k1_gemm_in(const float* __restrict__ x,
                                                  const _Float16* __restrict__ Wh_in,
                                                  const float* __restrict__ b_in,
                                                  float* __restrict__ inp) {
  __shared__ _Float16 Bs[TWO_H * 32];   // 16 KB: W_in[n][k0..k0+31] as f16
  const int tid  = threadIdx.x;
  const int lane = tid & 31;
  const int w    = tid >> 5;
  const int n16  = lane & 15;
  const int hi   = lane >> 4;
  const size_t m0   = (size_t)blockIdx.x * 128;
  const size_t arow = m0 + (size_t)w * 16 + n16;        // A row for this lane
  const float* __restrict__ ap = x + arow * TOKEN_DIM;
  const uint32_t bs_row = lds_off_of(Bs) + (uint32_t)tid * 64u;

  v8f acc[16];
#pragma unroll
  for (int j = 0; j < 16; ++j) acc[j] = (v8f)(0.0f);

  for (int ks = 0; ks < TOKEN_DIM / 32; ++ks) {
    const int k0 = ks * 32;
    __syncthreads();
#if USE_ASYNC_STAGE
    {   // stage Wh_in[n=tid][k0..k0+31] (64 B) via async global->LDS copies
      const uint64_t g_row =
          (uint64_t)(uintptr_t)(Wh_in + (size_t)tid * TOKEN_DIM + k0);
#pragma unroll
      for (int q = 0; q < 4; ++q)
        async_copy16(bs_row + (uint32_t)(q * 16), g_row + (uint64_t)(q * 16));
      wait_async0();
    }
#else
    {
      const _Float16* __restrict__ src = Wh_in + (size_t)tid * TOKEN_DIM + k0;
      _Float16* __restrict__ dst = &Bs[tid * 32];
#pragma unroll
      for (int q = 0; q < 32; ++q) dst[q] = src[q];
    }
#endif
    __syncthreads();

    const v16h a = frag_from_f32(ap + k0 + (hi ? 8 : 0));
#pragma unroll
    for (int j = 0; j < 16; ++j) {
      const v16h b = frag_from_h16(&Bs[(j * 16 + n16) * 32 + (hi ? 8 : 0)]);
      acc[j] = __builtin_amdgcn_wmma_f32_16x16x32_f16(
          false, a, false, b, (short)0, acc[j], false, false);
    }
  }

  // D layout: VGPR r -> row r + 8*hi ; lane&15 -> n within 16-tile
  const size_t mbase = m0 + (size_t)w * 16 + hi * 8;
#pragma unroll
  for (int j = 0; j < 16; ++j) {
    const int n = j * 16 + n16;
    const float bi = b_in[n];
#pragma unroll
    for (int r = 0; r < 8; ++r)
      inp[(mbase + r) * TWO_H + n] = acc[j][r] + bi;
  }
}

// =======================================================================
// K2: per (b, chunk, h) fold -> chunk summary (A, B) complex
// =======================================================================
__global__ __launch_bounds__(256) void k2_chunk(const float* __restrict__ inp,
                                                const unsigned char* __restrict__ mask,
                                                const float* __restrict__ log_radius,
                                                const float* __restrict__ phase,
                                                float* __restrict__ Ar, float* __restrict__ Ai,
                                                float* __restrict__ Br, float* __restrict__ Bi) {
  const int g = blockIdx.x * 256 + threadIdx.x;     // 0 .. NSUM-1
  const int h = g & (HDIM - 1);
  const int cidx = g >> 7;                          // b*NCHUNK + c
  const int b = cidx >> 7;
  const int c = cidx & (NCHUNK - 1);
  float lre, lim; lam_of(h, log_radius, phase, lre, lim);

  float ar = 1.f, ai = 0.f, br = 0.f, bi = 0.f;
  const size_t bt0 = (size_t)b * TSEQ + (size_t)c * CHUNK;
  for (int t = 0; t < CHUNK; ++t) {
    const size_t bt = bt0 + t;
    if (mask[bt]) {
      const float ur = inp[bt * TWO_H + h];
      const float ui = inp[bt * TWO_H + HDIM + h];
      const float nar = lre * ar - lim * ai, nai = lre * ai + lim * ar;
      const float nbr = lre * br - lim * bi + ur, nbi = lre * bi + lim * br + ui;
      ar = nar; ai = nai; br = nbr; bi = nbi;
    }
  }
  const size_t idx = (size_t)cidx * HDIM + h;
  Ar[idx] = ar; Ai[idx] = ai; Br[idx] = br; Bi[idx] = bi;
}

// =======================================================================
// K3: exclusive scan over chunks per (b,h) -> carry-in state per chunk
// =======================================================================
__global__ __launch_bounds__(256) void k3_scan(const float* __restrict__ Ar, const float* __restrict__ Ai,
                                               const float* __restrict__ Br, const float* __restrict__ Bi,
                                               float* __restrict__ Cr, float* __restrict__ Ci) {
  const int g = blockIdx.x * 256 + threadIdx.x;     // 0 .. 1023
  const int h = g & (HDIM - 1);
  const int b = g >> 7;
  float sr = 0.f, si = 0.f;
  for (int c = 0; c < NCHUNK; ++c) {
    const size_t idx = ((size_t)b * NCHUNK + c) * HDIM + h;
    Cr[idx] = sr; Ci[idx] = si;
    const float ar = Ar[idx], ai = Ai[idx], br = Br[idx], bi = Bi[idx];
    const float nr = ar * sr - ai * si + br;
    const float ni = ar * si + ai * sr + bi;
    sr = nr; si = ni;
  }
}

// =======================================================================
// K4: per (b,chunk): local scan w/ carry -> feat (LDS f16)
//     -> GEMM2 (WMMA f16, W_out async-staged to LDS) -> +b_out -> normalize
// =======================================================================
__global__ __launch_bounds__(256) void k4_final(const float* __restrict__ inp,
                                                const unsigned char* __restrict__ mask,
                                                const float* __restrict__ log_radius,
                                                const float* __restrict__ phase,
                                                const _Float16* __restrict__ Wh_out,
                                                const float* __restrict__ b_out,
                                                const float* __restrict__ Cr,
                                                const float* __restrict__ Ci,
                                                float* __restrict__ outg) {
  __shared__ _Float16 feat[CHUNK * TWO_H];   // 32 KB
  __shared__ _Float16 Bs[STATE_DIM * 32];    // 16 KB
  __shared__ float rnorm[CHUNK];             // 256 B
  const int tid = threadIdx.x;
  const int b = blockIdx.x >> 7;
  const int c = blockIdx.x & (NCHUNK - 1);
  const size_t bt0 = (size_t)b * TSEQ + (size_t)c * CHUNK;
  const uint32_t bs_row = lds_off_of(Bs) + (uint32_t)tid * 64u;

  // ---- phase 1: local scan with carry-in ----
  if (tid < HDIM) {
    const int h = tid;
    float lre, lim; lam_of(h, log_radius, phase, lre, lim);
    const size_t ci = ((size_t)b * NCHUNK + c) * HDIM + h;
    float hr = Cr[ci], him = Ci[ci];
    for (int t = 0; t < CHUNK; ++t) {
      const size_t bt = bt0 + t;
      if (mask[bt]) {
        const float ur = inp[bt * TWO_H + h];
        const float ui = inp[bt * TWO_H + HDIM + h];
        const float nr = lre * hr - lim * him + ur;
        const float ni = lre * him + lim * hr + ui;
        hr = nr; him = ni;
      }
      feat[t * TWO_H + h]        = (_Float16)hr;
      feat[t * TWO_H + HDIM + h] = (_Float16)him;
    }
  }
  __syncthreads();

  // ---- phase 2: GEMM2  out[64 x 256] = feat @ W_out^T ----
  const int lane = tid & 31, w = tid >> 5;
  const int n16 = lane & 15, hi = lane >> 4;
  const int mt = w & 3;       // m tile (16 rows)
  const int nb = w >> 2;      // n half (128 cols)
  v8f acc[8];
#pragma unroll
  for (int j = 0; j < 8; ++j) acc[j] = (v8f)(0.0f);

  for (int ks = 0; ks < TWO_H / 32; ++ks) {
    const int k0 = ks * 32;
    __syncthreads();
#if USE_ASYNC_STAGE
    {   // stage Wh_out[n=tid][k0..k0+31] via async global->LDS copies
      const uint64_t g_row =
          (uint64_t)(uintptr_t)(Wh_out + (size_t)tid * TWO_H + k0);
#pragma unroll
      for (int q = 0; q < 4; ++q)
        async_copy16(bs_row + (uint32_t)(q * 16), g_row + (uint64_t)(q * 16));
      wait_async0();
    }
#else
    {
      const _Float16* __restrict__ src = Wh_out + (size_t)tid * TWO_H + k0;
      _Float16* __restrict__ dst = &Bs[tid * 32];
#pragma unroll
      for (int q = 0; q < 32; ++q) dst[q] = src[q];
    }
#endif
    __syncthreads();

    const v16h a = frag_from_h16(&feat[(mt * 16 + n16) * TWO_H + k0 + (hi ? 8 : 0)]);
#pragma unroll
    for (int j = 0; j < 8; ++j) {
      const v16h bf = frag_from_h16(&Bs[(nb * 128 + j * 16 + n16) * 32 + (hi ? 8 : 0)]);
      acc[j] = __builtin_amdgcn_wmma_f32_16x16x32_f16(
          false, a, false, bf, (short)0, acc[j], false, false);
    }
  }
  __syncthreads();

  // ---- phase 3: add bias, row norms via LDS atomics, scale, store ----
  if (tid < CHUNK) rnorm[tid] = 0.f;
  __syncthreads();

  float bvals[8];
#pragma unroll
  for (int j = 0; j < 8; ++j) bvals[j] = b_out[nb * 128 + j * 16 + n16];

#pragma unroll
  for (int j = 0; j < 8; ++j) {
#pragma unroll
    for (int r = 0; r < 8; ++r) {
      const float v = acc[j][r] + bvals[j];
      atomicAdd(&rnorm[mt * 16 + 8 * hi + r], v * v);
    }
  }
  __syncthreads();
  if (tid < CHUNK) rnorm[tid] = 1.0f / fmaxf(sqrtf(rnorm[tid]), 1e-12f);
  __syncthreads();

#pragma unroll
  for (int j = 0; j < 8; ++j) {
    const int n = nb * 128 + j * 16 + n16;
#pragma unroll
    for (int r = 0; r < 8; ++r) {
      const int m = mt * 16 + 8 * hi + r;
      const float v = (acc[j][r] + bvals[j]) * rnorm[m];
      outg[(bt0 + m) * STATE_DIM + n] = v;
    }
  }
}

// =======================================================================
extern "C" void kernel_launch(void* const* d_in, const int* in_sizes, int n_in,
                              void* d_out, int out_size, void* d_ws, size_t ws_size,
                              hipStream_t stream) {
  const float*         x          = (const float*)d_in[0];
  const unsigned char* mask       = (const unsigned char*)d_in[1];
  const float*         W_in       = (const float*)d_in[2];
  const float*         b_in       = (const float*)d_in[3];
  const float*         log_radius = (const float*)d_in[4];
  const float*         phase      = (const float*)d_in[5];
  const float*         W_out      = (const float*)d_in[6];
  const float*         b_out      = (const float*)d_in[7];
  float*               out        = (float*)d_out;

  float* ws = (float*)d_ws;
  float* inp = ws;                                  // NTOK*TWO_H f32 (64 MB)
  size_t off = (size_t)NTOK * TWO_H;
  float* Ar = ws + off; off += NSUM;
  float* Ai = ws + off; off += NSUM;
  float* Br = ws + off; off += NSUM;
  float* Bi = ws + off; off += NSUM;
  float* Cr = ws + off; off += NSUM;
  float* Ci = ws + off; off += NSUM;
  _Float16* Wh_in  = (_Float16*)(ws + off);         // 256x512 f16
  _Float16* Wh_out = Wh_in + (size_t)TWO_H * TOKEN_DIM; // 256x256 f16

  k0_cvt_weights<<<(TWO_H * TOKEN_DIM + STATE_DIM * TWO_H) / 256, 256, 0, stream>>>(
      W_in, W_out, Wh_in, Wh_out);
  k1_gemm_in<<<NTOK / 128, 256, 0, stream>>>(x, Wh_in, b_in, inp);
  k2_chunk<<<NSUM / 256, 256, 0, stream>>>(inp, mask, log_radius, phase, Ar, Ai, Br, Bi);
  k3_scan<<<(BATCH * HDIM) / 256, 256, 0, stream>>>(Ar, Ai, Br, Bi, Cr, Ci);
  k4_final<<<BATCH * NCHUNK, 256, 0, stream>>>(inp, mask, log_radius, phase,
                                               Wh_out, b_out, Cr, Ci, out);
}
